// ViT_61375082660541
// MI455X (gfx1250) — matrix operand order repaired and checked by reading, not compiled, not run on hardware
//
#include <hip/hip_runtime.h>
#include <hip/hip_bf16.h>

// ---------------------------------------------------------------------------
// MI455X (gfx1250) ViT forward.  Memory-bound: ~1.32 GB of fp32 weights per
// pass -> ~57us floor at 23.3 TB/s.  All big GEMMs use v_wmma_f32_16x16x32_f16
// (wave32), streaming weights once; split-K partials give enough concurrent
// weight streams (>=2000 waves) to saturate HBM, at ~3% extra traffic.
// ---------------------------------------------------------------------------

typedef __attribute__((ext_vector_type(16))) _Float16 v16h;
typedef __attribute__((ext_vector_type(8)))  _Float16 v8h;
typedef __attribute__((ext_vector_type(8)))  float    v8f;

#define WMMA_F16(A_, B_, C_) \
  __builtin_amdgcn_wmma_f32_16x16x32_f16(false, (A_), false, (B_), (short)0, (C_), false, false)

// A fragment (16x32 f16, M x K). lane: m = l%16, half = l/16.
// element e -> K = half*16 + e  (contiguous, matches B loaders)
__device__ __forceinline__ v16h load_a16(const _Float16* __restrict__ p, int half) {
  v8h lo = *(const v8h*)(p + half * 16);       // K = half*16 .. +7
  v8h hi = *(const v8h*)(p + half * 16 + 8);   // K = half*16+8 .. +7
  v16h r;
#pragma unroll
  for (int i = 0; i < 8; ++i) { r[i] = lo[i]; r[8 + i] = hi[i]; }
  return r;
}

// B fragment (32x16 f16, K x N), weight row-major over K (W[n*ldw + k]).
// lane: n = l%16, half = l/16; element e -> K = half*16 + e  (contiguous!)
__device__ __forceinline__ v16h load_b16_nk(const float* __restrict__ p, int half) {
  const float4* q = (const float4*)(p + half * 16);
  float4 f0 = q[0], f1 = q[1], f2 = q[2], f3 = q[3];
  v16h r;
  r[0]  = (_Float16)f0.x; r[1]  = (_Float16)f0.y; r[2]  = (_Float16)f0.z; r[3]  = (_Float16)f0.w;
  r[4]  = (_Float16)f1.x; r[5]  = (_Float16)f1.y; r[6]  = (_Float16)f1.z; r[7]  = (_Float16)f1.w;
  r[8]  = (_Float16)f2.x; r[9]  = (_Float16)f2.y; r[10] = (_Float16)f2.z; r[11] = (_Float16)f2.w;
  r[12] = (_Float16)f3.x; r[13] = (_Float16)f3.y; r[14] = (_Float16)f3.z; r[15] = (_Float16)f3.w;
  return r;
}

// B fragment, weight K-major (W[k*ldb + n]) -- wqkv case.  Each k-row read is
// still lane-coalesced (16 consecutive floats per half-wave).
__device__ __forceinline__ v16h load_b16_km(const float* __restrict__ p, int half, long ldb) {
  v16h r;
#pragma unroll
  for (int e = 0; e < 16; ++e) r[e] = (_Float16)p[(long)(half * 16 + e) * ldb];
  return r;
}

// ---------------------------------------------------------------------------
// 1) Patch embedding + class token + position:  tok[b, s, d]  (f32 [32,65,128])
// ---------------------------------------------------------------------------
__global__ void patch_embed_kernel(const float* __restrict__ x, const float* __restrict__ cw,
                                   const float* __restrict__ cb, const float* __restrict__ cls,
                                   const float* __restrict__ pos, float* __restrict__ tok) {
  __shared__ float patch[768];
  int bp = blockIdx.x;            // b*65 + p
  int b = bp / 65, p = bp % 65;
  int d = threadIdx.x;            // 0..127
  if (p < 64) {
    int py = p >> 3, px = p & 7;
    for (int i = d; i < 768; i += 128) {
      int c = i >> 8, r = i & 255, ky = r >> 4, kx = r & 15;
      patch[i] = x[(((long)b * 3 + c) * 128 + (py * 16 + ky)) * 128 + (px * 16 + kx)];
    }
    __syncthreads();
    float acc = cb[d];
#pragma unroll 8
    for (int i = 0; i < 768; ++i) acc += patch[i] * cw[d * 768 + i];
    tok[(long)bp * 128 + d] = acc + pos[p * 128 + d];
  } else {
    tok[(long)bp * 128 + d] = cls[d] + pos[p * 128 + d];
  }
}

// ---------------------------------------------------------------------------
// 2) LayerNorm over last dim (128) -> f16 output (GEMM A panel)
// ---------------------------------------------------------------------------
__global__ void layernorm_f16_kernel(const float* __restrict__ X, const float* __restrict__ g,
                                     const float* __restrict__ be, _Float16* __restrict__ Y) {
  __shared__ float red[128];
  int row = blockIdx.x, tid = threadIdx.x;
  float v = X[(long)row * 128 + tid];
  red[tid] = v; __syncthreads();
  for (int s = 64; s > 0; s >>= 1) { if (tid < s) red[tid] += red[tid + s]; __syncthreads(); }
  float mean = red[0] * (1.0f / 128.0f);
  __syncthreads();
  float dv = v - mean;
  red[tid] = dv * dv; __syncthreads();
  for (int s = 64; s > 0; s >>= 1) { if (tid < s) red[tid] += red[tid + s]; __syncthreads(); }
  float var = red[0] * (1.0f / 128.0f);
  float y = dv * rsqrtf(var + 1e-5f) * g[tid] + be[tid];
  Y[(long)row * 128 + tid] = (_Float16)y;
}

// ---------------------------------------------------------------------------
// 3) QKV projection: per (h,k3) GEMM [32 x 2080] x [2080 x 2080] (K-major W).
//    A cols for k-tile t are n1[:, t*128 + h*32 .. +31].  780 wave jobs.
// ---------------------------------------------------------------------------
__global__ void gemm_qkv_kernel(const _Float16* __restrict__ A, const float* __restrict__ Wq,
                                float* __restrict__ qkv) {
  int wid = threadIdx.x >> 5;
  int job = blockIdx.x * (blockDim.x >> 5) + wid;
  if (job >= 780) return;                 // uniform per wave (EXEC stays all-1)
  int hk = job / 65, jn = job % 65;
  int h = hk & 3, k3 = hk >> 2;
  int lane = threadIdx.x & 31;
  int nl = lane & 15, half = lane >> 4;
  int n0 = jn * 32;
  const float* Wb = Wq + (long)h * 12979200 + (long)k3 * 4326400;  // [2080 x 2080], K-major
  v8f c00 = {}, c01 = {}, c10 = {}, c11 = {};
  const _Float16* ar0 = A + (long)nl * 8320;
  const _Float16* ar1 = A + (long)(nl + 16) * 8320;
  for (int t = 0; t < 65; ++t) {
    int ac = t * 128 + h * 32;
    const float* wb = Wb + (long)(t * 32) * 2080 + n0 + nl;
    __builtin_prefetch(wb + 8 * 2080, 0, 0);
    v16h a0 = load_a16(ar0 + ac, half);
    v16h a1 = load_a16(ar1 + ac, half);
    v16h b0 = load_b16_km(wb, half, 2080);
    v16h b1 = load_b16_km(wb + 16, half, 2080);
    c00 = WMMA_F16(a0, b0, c00);
    c01 = WMMA_F16(a0, b1, c01);
    c10 = WMMA_F16(a1, b0, c10);
    c11 = WMMA_F16(a1, b1, c11);
  }
  // D layout: elem r -> m = half*8 + r, n = nl  (per m-tile)
  float* Cb = qkv + (long)k3 * 266240 + (long)h * 2080;  // qkv[k3][b][h][sq]
#pragma unroll
  for (int r = 0; r < 8; ++r) {
    int m = half * 8 + r;
    int n = n0 + nl;
    Cb[(long)m * 8320 + n]        = c00[r];
    Cb[(long)m * 8320 + n + 16]   = c01[r];
    Cb[(long)(m + 16) * 8320 + n]      = c10[r];
    Cb[(long)(m + 16) * 8320 + n + 16] = c11[r];
  }
}

// ---------------------------------------------------------------------------
// 4) Attention per (b,h): softmax(q k^T), z = (p v)/sqrt(32), x2 = z + tok
// ---------------------------------------------------------------------------
__global__ void attention_kernel(const float* __restrict__ qkv, const float* __restrict__ tok,
                                 float* __restrict__ x2) {
  __shared__ float kf[65 * 32];
  __shared__ float vf[65 * 32];
  __shared__ float sc[65 * 65];
  int bh = blockIdx.x; int b = bh >> 2; int h = bh & 3;
  long base = (long)b * 8320 + (long)h * 2080;
  const float* qp = qkv + base;
  const float* kp = qkv + 266240 + base;
  const float* vp = qkv + 532480 + base;
  for (int i = threadIdx.x; i < 2080; i += blockDim.x) { kf[i] = kp[i]; vf[i] = vp[i]; }
  __syncthreads();
  int i = threadIdx.x;
  if (i < 65) {
    float q[32];
#pragma unroll
    for (int qq = 0; qq < 32; ++qq) q[qq] = qp[i * 32 + qq];
    float mx = -1e30f;
    for (int j = 0; j < 65; ++j) {
      float s = 0.f;
#pragma unroll
      for (int qq = 0; qq < 32; ++qq) s += q[qq] * kf[j * 32 + qq];
      sc[i * 65 + j] = s; mx = fmaxf(mx, s);
    }
    float sum = 0.f;
    for (int j = 0; j < 65; ++j) { float e = __expf(sc[i * 65 + j] - mx); sc[i * 65 + j] = e; sum += e; }
    float inv = 1.0f / sum;
    float z[32];
#pragma unroll
    for (int qq = 0; qq < 32; ++qq) z[qq] = 0.f;
    for (int j = 0; j < 65; ++j) {
      float p = sc[i * 65 + j] * inv;
#pragma unroll
      for (int qq = 0; qq < 32; ++qq) z[qq] += p * vf[j * 32 + qq];
    }
    long o = (long)b * 8320 + (long)i * 128 + h * 32;
#pragma unroll
    for (int qq = 0; qq < 32; ++qq)
      x2[o + qq] = z[qq] * 0.17677669529663687f + tok[o + qq];   // 1/sqrt(32) AFTER softmax
  }
}

// ---------------------------------------------------------------------------
// 5) Generic streaming WMMA GEMM: C[32 x N] = A_f16[32 x K] * W_f32[N x K]^T
//    N-major weight rows (k contiguous).  Split-K over gridDim.y writes f32
//    partials (epilogue deferred to combine_kernel); single-split launches may
//    fuse bias/ReLU/residual/f16-out directly.
// ---------------------------------------------------------------------------
__global__ void gemm_nk_kernel(const _Float16* __restrict__ A, int lda,
                               const float* __restrict__ W, int ldw,
                               const float* __restrict__ bias,
                               const float* __restrict__ resid,
                               float* __restrict__ Cf32,
                               _Float16* __restrict__ Cf16,
                               int ldc, int N, int kTilesPer, long splitStride, int relu) {
  int wid = threadIdx.x >> 5;
  int job = blockIdx.x * (blockDim.x >> 5) + wid;
  if (job >= (N >> 5)) return;            // uniform per wave
  int lane = threadIdx.x & 31;
  int nl = lane & 15, half = lane >> 4;
  int n0 = job * 32;
  int kStart = blockIdx.y * kTilesPer;
  if (Cf32) Cf32 += (long)blockIdx.y * splitStride;
  v8f c00 = {}, c01 = {}, c10 = {}, c11 = {};
  const _Float16* ar0 = A + (long)nl * lda + kStart * 32;
  const _Float16* ar1 = A + (long)(nl + 16) * lda + kStart * 32;
  const float* wr0 = W + (long)(n0 + nl) * ldw + kStart * 32;
  const float* wr1 = W + (long)(n0 + 16 + nl) * ldw + kStart * 32;
  for (int t = 0; t < kTilesPer; ++t) {
    int kc = t * 32;
    __builtin_prefetch(wr0 + kc + 256, 0, 0);   // global_prefetch_b8, ~1KB ahead
    __builtin_prefetch(wr1 + kc + 256, 0, 0);
    v16h a0 = load_a16(ar0 + kc, half);
    v16h a1 = load_a16(ar1 + kc, half);
    v16h b0 = load_b16_nk(wr0 + kc, half);
    v16h b1 = load_b16_nk(wr1 + kc, half);
    c00 = WMMA_F16(a0, b0, c00);
    c01 = WMMA_F16(a0, b1, c01);
    c10 = WMMA_F16(a1, b0, c10);
    c11 = WMMA_F16(a1, b1, c11);
  }
  float bs0 = bias ? bias[n0 + nl] : 0.f;
  float bs1 = bias ? bias[n0 + 16 + nl] : 0.f;
#pragma unroll
  for (int r = 0; r < 8; ++r) {
    int m = half * 8 + r;
    float v0 = c00[r] + bs0, v1 = c01[r] + bs1;
    float v2 = c10[r] + bs0, v3 = c11[r] + bs1;
    if (relu) { v0 = fmaxf(v0, 0.f); v1 = fmaxf(v1, 0.f); v2 = fmaxf(v2, 0.f); v3 = fmaxf(v3, 0.f); }
    long i0 = (long)m * ldc + n0 + nl;
    long i1 = i0 + 16;
    long i2 = (long)(m + 16) * ldc + n0 + nl;
    long i3 = i2 + 16;
    if (resid) { v0 += resid[i0]; v1 += resid[i1]; v2 += resid[i2]; v3 += resid[i3]; }
    if (Cf32) { Cf32[i0] = v0; Cf32[i1] = v1; Cf32[i2] = v2; Cf32[i3] = v3; }
    if (Cf16) { Cf16[i0] = (_Float16)v0; Cf16[i1] = (_Float16)v1;
                Cf16[i2] = (_Float16)v2; Cf16[i3] = (_Float16)v3; }
  }
}

// ---------------------------------------------------------------------------
// 6) Split-K combine: out = [relu](sum_s part[s] + bias[n]) [+ resid]
//    chunk = 32*N elements; i -> (m = i/N, n = i%N).
// ---------------------------------------------------------------------------
__global__ void combine_kernel(const float* __restrict__ part, int nsplit, long chunk,
                               const float* __restrict__ bias,
                               const float* __restrict__ resid,
                               float* __restrict__ outf32, _Float16* __restrict__ outf16,
                               int N, int relu) {
  long i = (long)blockIdx.x * blockDim.x + threadIdx.x;
  if (i >= chunk) return;
  int n = (int)(i % N);
  float s = bias ? bias[n] : 0.f;
  for (int t = 0; t < nsplit; ++t) s += part[(long)t * chunk + i];
  if (relu) s = fmaxf(s, 0.f);
  if (resid) s += resid[i];
  if (outf32) outf32[i] = s;
  if (outf16) outf16[i] = (_Float16)s;
}

// ---------------------------------------------------------------------------
// 7) Final 128->5:  out[b,c] = hh[b,:] . mlp_w2[c,:] + mlp_b2[c]
// ---------------------------------------------------------------------------
__global__ void final_head_kernel(const float* __restrict__ hh, const float* __restrict__ w2,
                                  const float* __restrict__ b2, float* __restrict__ out) {
  int t = blockIdx.x * blockDim.x + threadIdx.x;
  if (t >= 160) return;
  int b = t / 5, c = t % 5;
  float s = b2[c];
#pragma unroll 8
  for (int j = 0; j < 128; ++j) s += hh[b * 128 + j] * w2[c * 128 + j];
  out[t] = s;
}

// ---------------------------------------------------------------------------
extern "C" void kernel_launch(void* const* d_in, const int* in_sizes, int n_in,
                              void* d_out, int out_size, void* d_ws, size_t ws_size,
                              hipStream_t stream) {
  (void)in_sizes; (void)n_in; (void)out_size; (void)ws_size;
  const float* x         = (const float*)d_in[0];
  const float* conv_w    = (const float*)d_in[1];
  const float* conv_b    = (const float*)d_in[2];
  const float* class_hd  = (const float*)d_in[3];
  const float* position  = (const float*)d_in[4];
  const float* ln1_g     = (const float*)d_in[5];
  const float* ln1_b     = (const float*)d_in[6];
  const float* wqkv      = (const float*)d_in[7];
  const float* ln2_g     = (const float*)d_in[8];
  const float* ln2_b     = (const float*)d_in[9];
  const float* W1        = (const float*)d_in[10];
  const float* b1        = (const float*)d_in[11];
  const float* W2        = (const float*)d_in[12];
  const float* b2        = (const float*)d_in[13];
  const float* mlp_w1    = (const float*)d_in[14];
  const float* mlp_b1    = (const float*)d_in[15];
  const float* mlp_w2    = (const float*)d_in[16];
  const float* mlp_b2    = (const float*)d_in[17];
  float* out = (float*)d_out;

  // ---- workspace carve (256B aligned), total ~16.6 MB ----
  char* w = (char*)d_ws;
  auto carve = [&](size_t bytes) -> void* {
    void* p = (void*)w; w += (bytes + 255) & ~(size_t)255; return p;
  };
  float*    tok   = (float*)   carve(266240 * 4);      // [32,65,128]
  float*    x2    = (float*)   carve(266240 * 4);
  float*    qkv   = (float*)   carve(798720 * 4);      // [3][32][4][2080]
  _Float16* n1h   = (_Float16*)carve(266240 * 2);      // ln1 out, f16
  _Float16* n2h   = (_Float16*)carve(266240 * 2);      // ln2 out, f16
  _Float16* h1h   = (_Float16*)carve(532480 * 2);      // FFN hidden [32,16640] f16
  _Float16* fh    = (_Float16*)carve(266240 * 2);      // FFN out + resid, f16
  float*    partA = (float*)   carve(2129920 * 4);     // split-K partials (8.5MB, reused)
  float*    hh    = (float*)   carve(4096 * 4);        // head hidden [32,128]

  // 1) patch embed + class token + position
  patch_embed_kernel<<<32 * 65, 128, 0, stream>>>(x, conv_w, conv_b, class_hd, position, tok);
  // 2) ln1 -> f16
  layernorm_f16_kernel<<<2080, 128, 0, stream>>>(tok, ln1_g, ln1_b, n1h);
  // 3) QKV: 780 wave jobs, 2 waves/block -> 390 blocks
  gemm_qkv_kernel<<<390, 64, 0, stream>>>(n1h, wqkv, qkv);
  // 4) attention + residual -> x2
  attention_kernel<<<128, 128, 0, stream>>>(qkv, tok, x2);
  // 5) ln2 -> f16
  layernorm_f16_kernel<<<2080, 128, 0, stream>>>(x2, ln2_g, ln2_b, n2h);
  // 6) FFN1 [32,8320]x[8320,16640]: split-K=4 -> 2080 wave jobs (1040 blocks)
  gemm_nk_kernel<<<dim3(260, 4), 64, 0, stream>>>(n2h, 8320, W1, 8320, nullptr,
                                                  nullptr, partA, nullptr,
                                                  16640, 16640, 65, (long)532480, 0);
  combine_kernel<<<2080, 256, 0, stream>>>(partA, 4, (long)532480, b1, nullptr,
                                           nullptr, h1h, 16640, 1);
  // 7) FFN2 [32,16640]x[16640,8320]: split-K=8 -> 2080 wave jobs (1040 blocks)
  gemm_nk_kernel<<<dim3(130, 8), 64, 0, stream>>>(h1h, 16640, W2, 16640, nullptr,
                                                  nullptr, partA, nullptr,
                                                  8320, 8320, 65, (long)266240, 0);
  combine_kernel<<<1040, 256, 0, stream>>>(partA, 8, (long)266240, b2, x2,
                                           nullptr, fh, 8320, 0);
  // 8) head GEMM [32,8320]x[8320,128]: split-K=13 (20 k-tiles each)
  gemm_nk_kernel<<<dim3(2, 13), 64, 0, stream>>>(fh, 8320, mlp_w1, 8320, nullptr,
                                                 nullptr, partA, nullptr,
                                                 128, 128, 20, (long)4096, 0);
  combine_kernel<<<16, 256, 0, stream>>>(partA, 13, (long)4096, mlp_b1, nullptr,
                                         hh, nullptr, 128, 1);
  // 9) final 128->5
  final_head_kernel<<<1, 256, 0, stream>>>(hh, mlp_w2, mlp_b2, out);
}
